// VaryingHiddenLayer1000NeuronsNet_42236708389182
// MI455X (gfx1250) — compile-verified, matrix operand order checked by reading, not run on hardware
//
#include <hip/hip_runtime.h>
#include <hip/hip_bf16.h>

typedef _Float16 half_t;
typedef __attribute__((ext_vector_type(16))) _Float16 v16h;
typedef __attribute__((ext_vector_type(8)))  float    v8f;

// ---------------------------------------------------------------------------
// Fragment load: 16-bit A/B matrix layout for V_WMMA_F32_16X16X32_F16.
// Lane L = 16*half + r holds row (A) / column (B) index r, with K elements:
//   v16h[0..7]  = src[row][k0 + 8*half + 0..7]
//   v16h[8..15] = src[row][k0 + 16 + 8*half + 0..7]
// i.e. two contiguous 16-byte loads per lane.
// ---------------------------------------------------------------------------
union FragU { v16h v; float4 f4[2]; };

__device__ __forceinline__ v16h load_frag(const half_t* __restrict__ p) {
    FragU u;
    u.f4[0] = *(const float4*)(p);
    u.f4[1] = *(const float4*)(p + 16);
    return u.v;
}

// ---------------------------------------------------------------------------
// Fused GEMM (cur = inp @ W^T + b) + snntorch Leaky update (reset=subtract).
// Block: 256 threads = 8 waves, 64 (M) x 128 (N) tile.
// Wave: 2x2 grid of 16x16x32 WMMA tiles (32x32 per wave).
// A: [512, Kpad] f16 row-major.  W: [Npad>=grid cols, Kpad] f16 row-major.
// memState: f32 [512, 1024] persistent across t.
// outSpk/outMem: f32 [512, Nreal] slices of d_out for this timestep.
// actOut: f16 [512, 1024] spikes (padded cols stay zero), next layer's A.
// ---------------------------------------------------------------------------
__global__ __launch_bounds__(256)
void snn_gemm_leaky(const half_t* __restrict__ A, int lda,
                    const half_t* __restrict__ W, int ldw,
                    const float*  __restrict__ bias,
                    float*        __restrict__ memState,
                    float*        __restrict__ outSpk,
                    float*        __restrict__ outMem,
                    half_t*       __restrict__ actOut,
                    int Nreal, int Kpad)
{
    const int lane   = threadIdx.x & 31;
    const int waveId = threadIdx.x >> 5;
    const int hlf    = lane >> 4;      // 0 or 1
    const int r      = lane & 15;
    const int wm     = waveId >> 2;    // 0..1
    const int wn     = waveId & 3;     // 0..3
    const int rowBase = blockIdx.x * 64  + wm * 32;
    const int colBase = blockIdx.y * 128 + wn * 32;

    // Init accumulators with bias: C element (vgpr v, lane) has N = r (const
    // across v), so all 8 components share one bias value per n-tile.
    v8f acc[2][2];
    #pragma unroll
    for (int ni = 0; ni < 2; ++ni) {
        const int n = colBase + ni * 16 + r;
        const float bv = (n < Nreal) ? bias[n] : 0.0f;
        #pragma unroll
        for (int mi = 0; mi < 2; ++mi) {
            v8f c;
            #pragma unroll
            for (int v = 0; v < 8; ++v) c[v] = bv;
            acc[mi][ni] = c;
        }
    }

    const int koff = 8 * hlf;
    const size_t rA0 = (size_t)(rowBase + r)       * lda;
    const size_t rA1 = (size_t)(rowBase + 16 + r)  * lda;
    const size_t rB0 = (size_t)(colBase + r)       * ldw;
    const size_t rB1 = (size_t)(colBase + 16 + r)  * ldw;

    for (int k0 = 0; k0 < Kpad; k0 += 32) {
        const v16h a0 = load_frag(A + rA0 + k0 + koff);
        const v16h a1 = load_frag(A + rA1 + k0 + koff);
        const v16h b0 = load_frag(W + rB0 + k0 + koff);
        const v16h b1 = load_frag(W + rB1 + k0 + koff);
        acc[0][0] = __builtin_amdgcn_wmma_f32_16x16x32_f16(
            false, a0, false, b0, (short)0, acc[0][0], false, false);
        acc[0][1] = __builtin_amdgcn_wmma_f32_16x16x32_f16(
            false, a0, false, b1, (short)0, acc[0][1], false, false);
        acc[1][0] = __builtin_amdgcn_wmma_f32_16x16x32_f16(
            false, a1, false, b0, (short)0, acc[1][0], false, false);
        acc[1][1] = __builtin_amdgcn_wmma_f32_16x16x32_f16(
            false, a1, false, b1, (short)0, acc[1][1], false, false);
    }

    // Epilogue: leaky update. C layout: M = v + 8*half, N = r (within tile).
    #pragma unroll
    for (int mi = 0; mi < 2; ++mi) {
        #pragma unroll
        for (int ni = 0; ni < 2; ++ni) {
            const int n = colBase + ni * 16 + r;
            if (n >= Nreal) continue;
            #pragma unroll
            for (int v = 0; v < 8; ++v) {
                const int row = rowBase + mi * 16 + v + 8 * hlf;
                const size_t sidx = (size_t)row * 1024 + n;
                const float m_old = memState[sidx];
                const float cur   = acc[mi][ni][v];
                const float reset = (m_old > 1.0f) ? 1.0f : 0.0f;
                const float m_new = 0.9f * m_old + cur - reset;
                const float s     = (m_new > 1.0f) ? 1.0f : 0.0f;
                memState[sidx] = m_new;
                const size_t oidx = (size_t)row * Nreal + n;
                outMem[oidx] = m_new;
                outSpk[oidx] = s;
                if (actOut) actOut[sidx] = (half_t)s;
            }
        }
    }
}

// ---------------------------------------------------------------------------
// Convert + zero-pad fp32 [Nreal,Kreal] -> f16 [Npad,Kpad] row-major.
// Used for weights (once per call) and x_t (per timestep).
// ---------------------------------------------------------------------------
__global__ void cvt_pad_f16(const float* __restrict__ src, half_t* __restrict__ dst,
                            int Nreal, int Kreal, int Kpad, int total)
{
    const int idx = blockIdx.x * blockDim.x + threadIdx.x;
    if (idx >= total) return;
    const int n = idx / Kpad;
    const int k = idx - n * Kpad;
    const float v = (n < Nreal && k < Kreal) ? src[(size_t)n * Kreal + k] : 0.0f;
    dst[idx] = (half_t)v;
}

// Zero the membrane state (f32) and activation buffers (f16).
__global__ void zero_state(float* __restrict__ memf, int fn,
                           half_t* __restrict__ acth, int hn)
{
    const int idx = blockIdx.x * blockDim.x + threadIdx.x;
    if (idx < fn) memf[idx] = 0.0f;
    if (idx < hn) acth[idx] = (half_t)0.0f;
}

extern "C" void kernel_launch(void* const* d_in, const int* in_sizes, int n_in,
                              void* d_out, int out_size, void* d_ws, size_t ws_size,
                              hipStream_t stream) {
    (void)in_sizes; (void)n_in; (void)out_size; (void)ws_size;
    const float* x  = (const float*)d_in[0];
    const float* W0 = (const float*)d_in[1];
    const float* b0 = (const float*)d_in[2];
    const float* W1 = (const float*)d_in[3];
    const float* b1 = (const float*)d_in[4];
    const float* W2 = (const float*)d_in[5];
    const float* b2 = (const float*)d_in[6];
    const float* W3 = (const float*)d_in[7];
    const float* b3 = (const float*)d_in[8];
    float* out = (float*)d_out;

    // Workspace layout (bytes)
    char* ws = (char*)d_ws;
    half_t* W0h  = (half_t*)(ws + 0);          // [1024, 800]  f16
    half_t* W1h  = (half_t*)(ws + 1638400);    // [1024, 1024] f16
    half_t* W2h  = (half_t*)(ws + 3735552);    // [1024, 1024] f16
    half_t* W3h  = (half_t*)(ws + 5832704);    // [128, 1024]  f16
    half_t* xh   = (half_t*)(ws + 6094848);    // [512, 800]   f16
    half_t* act0 = (half_t*)(ws + 6914048);    // [512, 1024]  f16
    half_t* act1 = (half_t*)(ws + 7962624);
    half_t* act2 = (half_t*)(ws + 9011200);
    float*  mem  = (float*)(ws + 10059776);    // 4 x [512, 1024] f32

    // One-time per call: weight conversion + state zeroing.
    {
        int t0 = 1024 * 800;
        cvt_pad_f16<<<(t0 + 255) / 256, 256, 0, stream>>>(W0, W0h, 1000, 784, 800, t0);
        int t1 = 1024 * 1024;
        cvt_pad_f16<<<(t1 + 255) / 256, 256, 0, stream>>>(W1, W1h, 1000, 1000, 1024, t1);
        cvt_pad_f16<<<(t1 + 255) / 256, 256, 0, stream>>>(W2, W2h, 1000, 1000, 1024, t1);
        int t3 = 128 * 1024;
        cvt_pad_f16<<<(t3 + 255) / 256, 256, 0, stream>>>(W3, W3h, 10, 1000, 1024, t3);
        const int fn = 4 * 512 * 1024;            // mem floats
        const int hn = 3 * 512 * 1024;            // act halves
        zero_state<<<(fn + 255) / 256, 256, 0, stream>>>(mem, fn, act0, hn);
    }

    const size_t spkOff[4] = {0, 25600000, 51200000, 76800000};
    const size_t memOff[4] = {77056000, 102656000, 128256000, 153856000};
    const int memStride = 512 * 1024;

    for (int t = 0; t < 50; ++t) {
        // x_t fp32 [512,784] -> f16 [512,800]
        const int xt = 512 * 800;
        cvt_pad_f16<<<(xt + 255) / 256, 256, 0, stream>>>(
            x + (size_t)t * 512 * 784, xh, 512, 784, 800, xt);

        const size_t o01 = (size_t)t * 512 * 1000;
        const size_t o3  = (size_t)t * 512 * 10;

        // layer 0: [512,800] x [1000,800]^T
        snn_gemm_leaky<<<dim3(8, 8), 256, 0, stream>>>(
            xh, 800, W0h, 800, b0, mem + 0 * memStride,
            out + spkOff[0] + o01, out + memOff[0] + o01, act0, 1000, 800);
        // layer 1
        snn_gemm_leaky<<<dim3(8, 8), 256, 0, stream>>>(
            act0, 1024, W1h, 1024, b1, mem + 1 * memStride,
            out + spkOff[1] + o01, out + memOff[1] + o01, act1, 1000, 1024);
        // layer 2
        snn_gemm_leaky<<<dim3(8, 8), 256, 0, stream>>>(
            act1, 1024, W2h, 1024, b2, mem + 2 * memStride,
            out + spkOff[2] + o01, out + memOff[2] + o01, act2, 1000, 1024);
        // layer 3 (N=10)
        snn_gemm_leaky<<<dim3(8, 1), 256, 0, stream>>>(
            act2, 1024, W3h, 1024, b3, mem + 3 * memStride,
            out + spkOff[3] + o3, out + memOff[3] + o3, (half_t*)nullptr, 10, 1024);
    }
}
